// LinearProbeSvd_23132693856584
// MI455X (gfx1250) — matrix operand order repaired and checked by reading, not compiled, not run on hardware
//
#include <hip/hip_runtime.h>
#include <hip/hip_bf16.h>

typedef _Float16 f16;
typedef __attribute__((ext_vector_type(16))) _Float16 v16h;
typedef __attribute__((ext_vector_type(8)))  _Float16 v8h;
typedef __attribute__((ext_vector_type(8)))  float    v8f;

#define NROWS   65536   // 2 * B * N  (src rows then tgt rows)
#define HALFROWS 32768
#define DIN     1280
#define DMODEL  256

// ---------------- workspace layout (bytes) ----------------
#define WF16_OFF   ((size_t)0)                       // 1280*256 f16 (packed frags) = 655360
#define WQK_OFF    ((size_t)655360)                  // 256*512 f16 (packed frags)  = 262144
#define BIAS_OFF   ((size_t)917504)                  // 512 f32        = 2048
#define PROJ_OFF   ((size_t)919552)                  // 65536*256 f32  = 67108864
#define QK_OFF     ((size_t)68028416)                // 65536*512 f16  = 67108864
#define CONF_OFF   ((size_t)135137280)               // 65536 f32      = 262144
#define CORR_OFF   ((size_t)135399424)               // 65536*3 f32    = 786432

__device__ inline float redsum16(float v) {
  #pragma unroll
  for (int m = 1; m < 16; m <<= 1) v += __shfl_xor(v, m, 16);
  return v;
}
__device__ inline float redmax16(float v) {
  #pragma unroll
  for (int m = 1; m < 16; m <<= 1) v = fmaxf(v, __shfl_xor(v, m, 16));
  return v;
}
__device__ inline float redsum32(float v) {
  #pragma unroll
  for (int m = 1; m < 32; m <<= 1) v += __shfl_xor(v, m, 32);
  return v;
}
__device__ inline v16h cat8(v8h lo, v8h hi) {
  return __builtin_shufflevector(lo, hi, 0, 1, 2, 3, 4, 5, 6, 7,
                                 8, 9, 10, 11, 12, 13, 14, 15);
}

// -------- weight conversion + fragment packing --------
// Packed B-fragment layout: frag[((kt*NT + nt)*32 + lane)*16 + i] =
//   W[(kt*32 + (lane>=16?16:0) + i) * N + nt*16 + (lane&15)]
// so each lane reads its whole fragment as one aligned 32-byte v16h load.
__global__ void k_pack_wf(const float* __restrict__ Wf, f16* __restrict__ dst) {
  int idx = blockIdx.x * blockDim.x + threadIdx.x;
  if (idx >= DIN * DMODEL) return;
  const int i = idx & 15;
  const int lane = (idx >> 4) & 31;
  const int tile = idx >> 9;
  const int nt = tile & 15;        // 16 N-tiles
  const int kt = tile >> 4;        // 40 K-tiles
  const int k = kt * 32 + ((lane & 16) ? 16 : 0) + i;
  const int n = nt * 16 + (lane & 15);
  dst[idx] = (f16)Wf[k * DMODEL + n];
}

__global__ void k_pack_wqk(const float* __restrict__ Wq, const float* __restrict__ bq,
                           const float* __restrict__ Wk, const float* __restrict__ bk,
                           f16* __restrict__ dst, float* __restrict__ bias) {
  const float scale = 0.0625f; // 1/sqrt(256)
  int idx = blockIdx.x * blockDim.x + threadIdx.x;
  if (idx < DMODEL * 512) {
    const int i = idx & 15;
    const int lane = (idx >> 4) & 31;
    const int tile = idx >> 9;
    const int nt = tile & 31;      // 32 N-tiles
    const int kt = tile >> 5;      // 8 K-tiles
    const int k = kt * 32 + ((lane & 16) ? 16 : 0) + i;
    const int n = nt * 16 + (lane & 15);
    const float v = (n < DMODEL) ? Wq[k * DMODEL + n] * scale
                                 : Wk[k * DMODEL + (n - DMODEL)];
    dst[idx] = (f16)v;
  } else if (idx < DMODEL * 512 + 512) {
    const int j = idx - DMODEL * 512;
    bias[j] = (j < DMODEL) ? bq[j] * scale : bk[j - DMODEL];
  }
}

// -------- feat projection GEMM: 65536 x 1280 x 256, f16 WMMA, fp32 accum --------
// Double-buffered LDS A-tile, one barrier per K-step, next A prefetched during WMMAs.
__global__ void __launch_bounds__(128)
k_proj(const float* __restrict__ feat1, const float* __restrict__ feat2,
       const f16* __restrict__ wf16, const float* __restrict__ bf,
       float* __restrict__ proj) {
  __shared__ f16 Alds[2][16 * 32];
  const int tid = threadIdx.x;
  const int lane = tid & 31;
  const int wave = tid >> 5;               // 0..3, each wave owns 64 output cols
  const int rowbase = blockIdx.x * 16;
  const float* src = (rowbase < HALFROWS) ? (feat1 + (size_t)rowbase * DIN)
                                          : (feat2 + (size_t)(rowbase - HALFROWS) * DIN);
  const int e0 = tid * 4;                  // staging: 4 consecutive elems, one row
  const int sm = e0 >> 5;
  const int sk = e0 & 31;
  float av[4];
  #pragma unroll
  for (int j = 0; j < 4; ++j) av[j] = src[(size_t)sm * DIN + sk + j];

  v8f acc[4] = {};
  for (int kt = 0; kt < DIN / 32; ++kt) {
    f16* ab = &Alds[kt & 1][0];
    #pragma unroll
    for (int j = 0; j < 4; ++j) ab[e0 + j] = (f16)av[j];
    __syncthreads();
    if (kt + 1 < DIN / 32) {
      #pragma unroll
      for (int j = 0; j < 4; ++j)
        av[j] = src[(size_t)sm * DIN + (kt + 1) * 32 + sk + j];
    }
    const f16* ap = ab + (lane & 15) * 32 + ((lane >= 16) ? 16 : 0);
    v16h a = cat8(*(const v8h*)ap, *(const v8h*)(ap + 8));
    v16h bfrag[4];
    #pragma unroll
    for (int t = 0; t < 4; ++t)
      bfrag[t] = *(const v16h*)(wf16 + ((size_t)(kt * 16 + wave * 4 + t) * 32 + lane) * 16);
    #pragma unroll
    for (int t = 0; t < 4; ++t)
      acc[t] = __builtin_amdgcn_wmma_f32_16x16x32_f16(false, a, false, bfrag[t],
                                                      (short)0, acc[t], false, false);
  }
  const int mrow = (lane >= 16) ? 8 : 0;
  #pragma unroll
  for (int t = 0; t < 4; ++t) {
    const int n = wave * 64 + t * 16 + (lane & 15);
    #pragma unroll
    for (int r = 0; r < 8; ++r)
      proj[(size_t)(rowbase + mrow + r) * DMODEL + n] = acc[t][r] + bf[n];
  }
}

// -------- overlap confidence: sigmoid(proj @ Wc + bc), one warp per row --------
__global__ void __launch_bounds__(256)
k_conf(const float* __restrict__ proj, const float* __restrict__ Wc,
       const float* __restrict__ bc, float* __restrict__ conf) {
  const int lane = threadIdx.x & 31;
  const int row = blockIdx.x * 8 + (threadIdx.x >> 5);
  float s = 0.f;
  #pragma unroll
  for (int j = 0; j < 8; ++j) {
    int c = lane + 32 * j;
    s += proj[(size_t)row * DMODEL + c] * Wc[c];
  }
  s = redsum32(s);
  if (lane == 0) conf[row] = 1.f / (1.f + __expf(-(s + bc[0])));
}

// -------- sinusoidal positional embedding, added in place --------
__global__ void __launch_bounds__(256)
k_peadd(const float* __restrict__ xyz1, const float* __restrict__ xyz2,
        float* __restrict__ proj) {
  const int row = blockIdx.x;
  const int j = threadIdx.x;
  if (j >= 252) return;
  const float* xyz = (row < HALFROWS) ? (xyz1 + (size_t)row * 3)
                                      : (xyz2 + (size_t)(row - HALFROWS) * 3);
  const int c = j / 84, jj = j % 84, i = jj >> 1;
  // dim_t = 10000^(2i/84); pd = x*2pi / dim_t
  const float inv_t = __expf(-(2.f * (float)i / 84.f) * 9.210340371976184f);
  const float pd = xyz[c] * 6.283185307179586f * inv_t;
  const float v = (jj & 1) ? __cosf(pd) : __sinf(pd);
  proj[(size_t)row * DMODEL + j] += v;
}

// -------- q|k projection GEMM: 65536 x 256 x 512, output f16 --------
__global__ void __launch_bounds__(256)
k_qk(const float* __restrict__ proj, const f16* __restrict__ wqk16,
     const float* __restrict__ bias, f16* __restrict__ qk) {
  __shared__ f16 Alds[2][16 * 32];
  const int tid = threadIdx.x;
  const int lane = tid & 31;
  const int wave = tid >> 5;               // 0..7, each wave owns 64 of 512 cols
  const int rowbase = blockIdx.x * 16;
  const float* src = proj + (size_t)rowbase * DMODEL;
  const int e0 = tid * 2;
  const int sm = e0 >> 5;
  const int sk = e0 & 31;
  float av[2];
  #pragma unroll
  for (int j = 0; j < 2; ++j) av[j] = src[(size_t)sm * DMODEL + sk + j];

  v8f acc[4] = {};
  for (int kt = 0; kt < DMODEL / 32; ++kt) {
    f16* ab = &Alds[kt & 1][0];
    #pragma unroll
    for (int j = 0; j < 2; ++j) ab[e0 + j] = (f16)av[j];
    __syncthreads();
    if (kt + 1 < DMODEL / 32) {
      #pragma unroll
      for (int j = 0; j < 2; ++j)
        av[j] = src[(size_t)sm * DMODEL + (kt + 1) * 32 + sk + j];
    }
    const f16* ap = ab + (lane & 15) * 32 + ((lane >= 16) ? 16 : 0);
    v16h a = cat8(*(const v8h*)ap, *(const v8h*)(ap + 8));
    v16h bfrag[4];
    #pragma unroll
    for (int t = 0; t < 4; ++t)
      bfrag[t] = *(const v16h*)(wqk16 + ((size_t)(kt * 32 + wave * 4 + t) * 32 + lane) * 16);
    #pragma unroll
    for (int t = 0; t < 4; ++t)
      acc[t] = __builtin_amdgcn_wmma_f32_16x16x32_f16(false, a, false, bfrag[t],
                                                      (short)0, acc[t], false, false);
  }
  const int mrow = (lane >= 16) ? 8 : 0;
  #pragma unroll
  for (int t = 0; t < 4; ++t) {
    const int n = wave * 64 + t * 16 + (lane & 15);
    #pragma unroll
    for (int r = 0; r < 8; ++r)
      qk[(size_t)(rowbase + mrow + r) * 512 + n] = (f16)(acc[t][r] + bias[n]);
  }
}

// -------- flash attention with 3-dim value (xyz) --------
// task: dir (2) x batch (16) x qtile (128); one wave per task.
__global__ void __launch_bounds__(128)
k_attn(const f16* __restrict__ qk, const float* __restrict__ xyz1,
       const float* __restrict__ xyz2, float* __restrict__ corr) {
  const int lane = threadIdx.x & 31;
  const int task = blockIdx.x * 4 + (threadIdx.x >> 5);
  const int dir = task >> 11;
  const int rem = task & 2047;
  const int b = rem >> 7;
  const int qt = rem & 127;
  const size_t qrow_base = (size_t)dir * HALFROWS + (size_t)b * 2048 + qt * 16;
  const size_t krow_base = (size_t)(1 - dir) * HALFROWS + (size_t)b * 2048;
  const float* vsrc = (dir == 0) ? (xyz2 + (size_t)b * 2048 * 3)
                                 : (xyz1 + (size_t)b * 2048 * 3);
  // load the 16x256 f16 query tile (A fragments for 8 K-steps): 2x16B per step
  v16h qa[8];
  const f16* qp = qk + (qrow_base + (lane & 15)) * 512 + ((lane >= 16) ? 16 : 0);
  #pragma unroll
  for (int ks = 0; ks < 8; ++ks)
    qa[ks] = cat8(*(const v8h*)(qp + ks * 32), *(const v8h*)(qp + ks * 32 + 8));

  float rmax[8], rsum[8], racc[8][3];
  #pragma unroll
  for (int r = 0; r < 8; ++r) {
    rmax[r] = -3.0e38f; rsum[r] = 0.f;
    racc[r][0] = racc[r][1] = racc[r][2] = 0.f;
  }

  for (int kt2 = 0; kt2 < 128; ++kt2) {
    const int key = kt2 * 16 + (lane & 15);
    // key fragment is 32B contiguous per lane in the qk buffer
    const f16* kp = qk + (krow_base + key) * 512 + 256 + ((lane >= 16) ? 16 : 0);
    const float vx = vsrc[key * 3 + 0];
    const float vy = vsrc[key * 3 + 1];
    const float vz = vsrc[key * 3 + 2];
    v16h bm[8];
    #pragma unroll
    for (int ks = 0; ks < 8; ++ks) bm[ks] = *(const v16h*)(kp + ks * 32);
    v8f S = {};
    #pragma unroll
    for (int ks = 0; ks < 8; ++ks)
      S = __builtin_amdgcn_wmma_f32_16x16x32_f16(false, qa[ks], false, bm[ks],
                                                 (short)0, S, false, false);
    #pragma unroll
    for (int r = 0; r < 8; ++r) {
      float s = S[r];
      float tmax = redmax16(s);
      float newmax = fmaxf(rmax[r], tmax);
      float sc_old = __expf(rmax[r] - newmax);
      float p = __expf(s - newmax);
      float psum = redsum16(p);
      rsum[r] = rsum[r] * sc_old + psum;
      racc[r][0] = racc[r][0] * sc_old + redsum16(p * vx);
      racc[r][1] = racc[r][1] * sc_old + redsum16(p * vy);
      racc[r][2] = racc[r][2] * sc_old + redsum16(p * vz);
      rmax[r] = newmax;
    }
  }
  const int mrow = (lane >= 16) ? 8 : 0;
  const int c = lane & 15;
  if (c < 3) {
    #pragma unroll
    for (int r = 0; r < 8; ++r) {
      float inv = 1.f / fmaxf(rsum[r], 1e-30f);
      corr[(qrow_base + mrow + r) * 3 + c] = racc[r][c] * inv;
    }
  }
}

// -------- weighted Kabsch per batch: centroids, 3x3 cov, Jacobi SVD --------
__global__ void __launch_bounds__(256)
k_kabsch(const float* __restrict__ xyz1, const float* __restrict__ xyz2,
         const float* __restrict__ corr, const float* __restrict__ conf,
         float* __restrict__ out) {
  __shared__ float sd[256 * 12];
  const int b = blockIdx.x;
  const int tid = threadIdx.x;

  // phase 1: wsum, sum(w*a), sum(w*b)
  float part[7] = {};
  for (int i = tid; i < 4096; i += 256) {
    float ax, ay, az, bx, by, bz, w;
    if (i < 2048) {
      ax = xyz1[((size_t)b * 2048 + i) * 3 + 0];
      ay = xyz1[((size_t)b * 2048 + i) * 3 + 1];
      az = xyz1[((size_t)b * 2048 + i) * 3 + 2];
      bx = corr[((size_t)b * 2048 + i) * 3 + 0];
      by = corr[((size_t)b * 2048 + i) * 3 + 1];
      bz = corr[((size_t)b * 2048 + i) * 3 + 2];
      w  = conf[(size_t)b * 2048 + i];
    } else {
      int ii = i - 2048;
      ax = corr[((size_t)HALFROWS + b * 2048 + ii) * 3 + 0];
      ay = corr[((size_t)HALFROWS + b * 2048 + ii) * 3 + 1];
      az = corr[((size_t)HALFROWS + b * 2048 + ii) * 3 + 2];
      bx = xyz2[((size_t)b * 2048 + ii) * 3 + 0];
      by = xyz2[((size_t)b * 2048 + ii) * 3 + 1];
      bz = xyz2[((size_t)b * 2048 + ii) * 3 + 2];
      w  = conf[(size_t)HALFROWS + b * 2048 + ii];
    }
    part[0] += w;
    part[1] += w * ax; part[2] += w * ay; part[3] += w * az;
    part[4] += w * bx; part[5] += w * by; part[6] += w * bz;
  }
  for (int j = 0; j < 7; ++j) sd[tid * 8 + j] = part[j];
  for (int s = 128; s > 0; s >>= 1) {
    __syncthreads();
    if (tid < s) for (int j = 0; j < 7; ++j) sd[tid * 8 + j] += sd[(tid + s) * 8 + j];
  }
  __syncthreads();
  const float denom = fmaxf(sd[0], 1e-6f);
  const float cax = sd[1] / denom, cay = sd[2] / denom, caz = sd[3] / denom;
  const float cbx = sd[4] / denom, cby = sd[5] / denom, cbz = sd[6] / denom;
  __syncthreads();

  // phase 2: cov[3][3] = sum wn * (a-ca)(b-cb)^T
  float cv[9] = {};
  for (int i = tid; i < 4096; i += 256) {
    float ax, ay, az, bx, by, bz, w;
    if (i < 2048) {
      ax = xyz1[((size_t)b * 2048 + i) * 3 + 0];
      ay = xyz1[((size_t)b * 2048 + i) * 3 + 1];
      az = xyz1[((size_t)b * 2048 + i) * 3 + 2];
      bx = corr[((size_t)b * 2048 + i) * 3 + 0];
      by = corr[((size_t)b * 2048 + i) * 3 + 1];
      bz = corr[((size_t)b * 2048 + i) * 3 + 2];
      w  = conf[(size_t)b * 2048 + i];
    } else {
      int ii = i - 2048;
      ax = corr[((size_t)HALFROWS + b * 2048 + ii) * 3 + 0];
      ay = corr[((size_t)HALFROWS + b * 2048 + ii) * 3 + 1];
      az = corr[((size_t)HALFROWS + b * 2048 + ii) * 3 + 2];
      bx = xyz2[((size_t)b * 2048 + ii) * 3 + 0];
      by = xyz2[((size_t)b * 2048 + ii) * 3 + 1];
      bz = xyz2[((size_t)b * 2048 + ii) * 3 + 2];
      w  = conf[(size_t)HALFROWS + b * 2048 + ii];
    }
    const float wn = w / denom;
    const float dax = ax - cax, day = ay - cay, daz = az - caz;
    const float dbx = (bx - cbx) * wn, dby = (by - cby) * wn, dbz = (bz - cbz) * wn;
    cv[0] += dax * dbx; cv[1] += dax * dby; cv[2] += dax * dbz;
    cv[3] += day * dbx; cv[4] += day * dby; cv[5] += day * dbz;
    cv[6] += daz * dbx; cv[7] += daz * dby; cv[8] += daz * dbz;
  }
  for (int j = 0; j < 9; ++j) sd[tid * 12 + j] = cv[j];
  for (int s = 128; s > 0; s >>= 1) {
    __syncthreads();
    if (tid < s) for (int j = 0; j < 9; ++j) sd[tid * 12 + j] += sd[(tid + s) * 12 + j];
  }
  __syncthreads();

  if (tid == 0) {
    float Bc[3][3], V[3][3];
    for (int r = 0; r < 3; ++r)
      for (int c = 0; c < 3; ++c) {
        Bc[r][c] = sd[r * 3 + c];
        V[r][c] = (r == c) ? 1.f : 0.f;
      }
    // one-sided Jacobi: orthogonalize columns of Bc, accumulate V
    const int PP[3] = {0, 0, 1}, QQ[3] = {1, 2, 2};
    for (int it = 0; it < 36; ++it) {
      const int p = PP[it % 3], q = QQ[it % 3];
      float al = 0.f, be = 0.f, ga = 0.f;
      for (int r = 0; r < 3; ++r) {
        al += Bc[r][p] * Bc[r][p];
        be += Bc[r][q] * Bc[r][q];
        ga += Bc[r][p] * Bc[r][q];
      }
      if (fabsf(ga) > 1e-20f) {
        const float ze = (be - al) / (2.f * ga);
        const float t = ((ze >= 0.f) ? 1.f : -1.f) / (fabsf(ze) + sqrtf(1.f + ze * ze));
        const float cc = 1.f / sqrtf(1.f + t * t);
        const float ss = cc * t;
        for (int r = 0; r < 3; ++r) {
          float bp = Bc[r][p], bq2 = Bc[r][q];
          Bc[r][p] = cc * bp - ss * bq2; Bc[r][q] = ss * bp + cc * bq2;
          float vp = V[r][p], vq = V[r][q];
          V[r][p] = cc * vp - ss * vq; V[r][q] = ss * vp + cc * vq;
        }
      }
    }
    float sig[3];
    for (int c = 0; c < 3; ++c)
      sig[c] = sqrtf(Bc[0][c] * Bc[0][c] + Bc[1][c] * Bc[1][c] + Bc[2][c] * Bc[2][c]);
    // sort singular values descending, permute columns of Bc and V
    for (int a0 = 0; a0 < 2; ++a0)
      for (int a1 = a0 + 1; a1 < 3; ++a1)
        if (sig[a0] < sig[a1]) {
          float t0 = sig[a0]; sig[a0] = sig[a1]; sig[a1] = t0;
          for (int r = 0; r < 3; ++r) {
            float tb = Bc[r][a0]; Bc[r][a0] = Bc[r][a1]; Bc[r][a1] = tb;
            float tv = V[r][a0]; V[r][a0] = V[r][a1]; V[r][a1] = tv;
          }
        }
    float U[3][3];
    for (int c = 0; c < 3; ++c) {
      const float inv = 1.f / fmaxf(sig[c], 1e-12f);
      for (int r = 0; r < 3; ++r) U[r][c] = Bc[r][c] * inv;
    }
    float rot[3][3];
    for (int r = 0; r < 3; ++r)
      for (int c = 0; c < 3; ++c)
        rot[r][c] = V[r][0] * U[c][0] + V[r][1] * U[c][1] + V[r][2] * U[c][2];
    const float det =
        rot[0][0] * (rot[1][1] * rot[2][2] - rot[1][2] * rot[2][1]) -
        rot[0][1] * (rot[1][0] * rot[2][2] - rot[1][2] * rot[2][0]) +
        rot[0][2] * (rot[1][0] * rot[2][1] - rot[1][1] * rot[2][0]);
    if (det <= 0.f) {
      for (int r = 0; r < 3; ++r) V[r][2] = -V[r][2];
      for (int r = 0; r < 3; ++r)
        for (int c = 0; c < 3; ++c)
          rot[r][c] = V[r][0] * U[c][0] + V[r][1] * U[c][1] + V[r][2] * U[c][2];
    }
    const float tx = -(rot[0][0] * cax + rot[0][1] * cay + rot[0][2] * caz) + cbx;
    const float ty = -(rot[1][0] * cax + rot[1][1] * cay + rot[1][2] * caz) + cby;
    const float tz = -(rot[2][0] * cax + rot[2][1] * cay + rot[2][2] * caz) + cbz;
    for (int r = 0; r < 3; ++r)
      for (int c = 0; c < 3; ++c) out[b * 12 + r * 4 + c] = rot[r][c];
    out[b * 12 + 0 * 4 + 3] = tx;
    out[b * 12 + 1 * 4 + 3] = ty;
    out[b * 12 + 2 * 4 + 3] = tz;
  }
}

extern "C" void kernel_launch(void* const* d_in, const int* in_sizes, int n_in,
                              void* d_out, int out_size, void* d_ws, size_t ws_size,
                              hipStream_t stream) {
  const float* feat1 = (const float*)d_in[0];
  const float* xyz1  = (const float*)d_in[1];
  const float* feat2 = (const float*)d_in[2];
  const float* xyz2  = (const float*)d_in[3];
  const float* Wf    = (const float*)d_in[4];
  const float* bf    = (const float*)d_in[5];
  const float* Wq    = (const float*)d_in[6];
  const float* bq    = (const float*)d_in[7];
  const float* Wk    = (const float*)d_in[8];
  const float* bk    = (const float*)d_in[9];
  const float* Wc    = (const float*)d_in[10];
  const float* bc    = (const float*)d_in[11];
  float* out = (float*)d_out;

  char* ws = (char*)d_ws;
  f16*   wf16  = (f16*)(ws + WF16_OFF);
  f16*   wqk16 = (f16*)(ws + WQK_OFF);
  float* biasq = (float*)(ws + BIAS_OFF);
  float* proj  = (float*)(ws + PROJ_OFF);
  f16*   qk    = (f16*)(ws + QK_OFF);
  float* conf  = (float*)(ws + CONF_OFF);
  float* corr  = (float*)(ws + CORR_OFF);

  k_pack_wf<<<(DIN * DMODEL + 255) / 256, 256, 0, stream>>>(Wf, wf16);
  k_pack_wqk<<<(DMODEL * 512 + 512 + 255) / 256, 256, 0, stream>>>(Wq, bq, Wk, bk, wqk16, biasq);
  k_proj<<<NROWS / 16, 128, 0, stream>>>(feat1, feat2, wf16, bf, proj);
  k_conf<<<NROWS / 8, 256, 0, stream>>>(proj, Wc, bc, conf);
  k_peadd<<<NROWS, 256, 0, stream>>>(xyz1, xyz2, proj);
  k_qk<<<NROWS / 16, 256, 0, stream>>>(proj, wqk16, biasq, qk);
  k_attn<<<4096 / 4, 128, 0, stream>>>(qk, xyz1, xyz2, corr);
  k_kabsch<<<16, 256, 0, stream>>>(xyz1, xyz2, corr, conf, out);
}
